// ChildSumTreeLSTM_35940286333320
// MI455X (gfx1250) — matrix operand order repaired and checked by reading, not compiled, not run on hardware
//
#include <hip/hip_runtime.h>
#include <hip/hip_bf16.h>
#include <math.h>
#include <stdint.h>

// ---------------------------------------------------------------------------
// ChildSum Tree-LSTM on gfx1250 (MI455X), wave32 + v_wmma_f32_16x16x32_bf16,
// LDS-staged B panels via GLOBAL_LOAD_ASYNC_TO_LDS (ASYNCcnt, inline asm).
// Tree is a reversed complete 4-ary heap: node id = 4095 - heap_j, so levels
// (bottom-up) are fixed id ranges:
//   leaves: [0,3072)   internal: [3072,3755) [3755,4011) [4011,4075)
//                                 [4075,4091) [4091,4095) [4095,4096)
// ---------------------------------------------------------------------------

#define NN   4096
#define DM   512          // MEM_DIM == IN_DIM == 512
#define KCH  4

typedef __attribute__((ext_vector_type(16))) __bf16 v16bf;
typedef __attribute__((ext_vector_type(8)))  __bf16 v8bf;
typedef __attribute__((ext_vector_type(8)))  float  v8f;

#if defined(__has_builtin) && __has_builtin(__builtin_amdgcn_s_wait_asynccnt)
#  define WAIT_ASYNC() __builtin_amdgcn_s_wait_asynccnt(0)
#else
#  define WAIT_ASYNC() asm volatile("s_wait_asynccnt 0x0" ::: "memory")
#endif

__device__ __forceinline__ float sigmoidf_(float x) {
    return 1.0f / (1.0f + __expf(-x));
}

// GLOBAL_LOAD_ASYNC_TO_LDS_B128 (VGLOBAL op 98): VDST = VGPR with LDS byte
// address, VADDR = 64-bit global address, SADDR = off. Tracked by ASYNCcnt.
__device__ __forceinline__ void async_load_b128_to_lds(const void* gsrc, void* lds_dst) {
    unsigned lds_off = (unsigned)(uintptr_t)(__attribute__((address_space(3))) void*)lds_dst;
    unsigned long long ga = (unsigned long long)(uintptr_t)gsrc;
    asm volatile("global_load_async_to_lds_b128 %0, %1, off"
                 :: "v"(lds_off), "v"(ga)
                 : "memory");
}

// Stage one 64col x 32K bf16 B panel (4KB) into LDS: 256 threads x 16 bytes.
// LDS layout: panel[col][kk] contiguous in kk -> a lane's 16-element fragment
// chunk is one 32-byte aligned LDS read.
__device__ __forceinline__ void load_B_panel(const __bf16* __restrict__ BT,
                                             int tn, int K, int k0,
                                             __bf16* lds_panel, int tid) {
    const int col   = tid >> 2;         // 0..63
    const int chunk = (tid & 3) * 8;    // K offset: 0,8,16,24
    const __bf16* src = BT + (size_t)(tn + col) * K + k0 + chunk;
    __bf16* dst = lds_panel + col * 32 + chunk;
    async_load_b128_to_lds((const void*)src, (void*)dst);
}

// ---------------------------------------------------------------------------
// WMMA GEMM:  C[Mpad x N] = A[Mpad x K](bf16,row-major) * B  + bias
// B supplied TRANSPOSED: BT[N x K] row-major. Block = 8 waves = 128x64 C tile;
// wave w owns rows [tm*16, tm*16+16) x 64 cols (4 accumulators, A reused 4x,
// B panel staged in LDS and reused by all 8 waves, double buffered).
// Tail waves (tm >= Mtiles) compute on clamped row-0 data unconditionally --
// keeping the k-loop free of divergent control flow (no saveexec / acc moves)
// -- and are predicated out only at the final store.
// ---------------------------------------------------------------------------
__global__ __launch_bounds__(256)
void wmma_gemm_bf16nt(const __bf16* __restrict__ A,
                      const __bf16* __restrict__ BT,
                      const float*  __restrict__ bias,
                      float* __restrict__ C,
                      int Mtiles, int Nstrips, int K, int N)
{
    __shared__ __attribute__((aligned(64))) __bf16 ldsB[2][64 * 32];

    const int tid  = threadIdx.x;
    const int wv   = tid >> 5;
    const int lane = tid & 31;
    const int half = lane >> 4;          // 0: lanes 0-15, 1: lanes 16-31
    const int l    = lane & 15;

    const int tn     = (blockIdx.x % Nstrips) * 64;
    const int tm     = (blockIdx.x / Nstrips) * 8 + wv;
    const bool active = (tm < Mtiles);   // wave-uniform

    v8f acc[4] = {};

    // A fragment row for this lane (per ISA 16-bit A layout: halves hold
    // different K groups of the same 16 rows). Clamped for tail waves.
    const __bf16* pA = A + (size_t)((active ? tm * 16 : 0) + l) * K;

    load_B_panel(BT, tn, K, 0, ldsB[0], tid);

    for (int k0 = 0; k0 < K; k0 += 32) {
        const int buf = (k0 >> 5) & 1;
        WAIT_ASYNC();                    // own async stores to LDS complete
        __syncthreads();                 // panel visible to all waves
        if (k0 + 32 < K)
            load_B_panel(BT, tn, K, k0 + 32, ldsB[buf ^ 1], tid);

        __builtin_prefetch(pA + k0 + 64, 0, 1);   // global_prefetch_b8 (A stream)
        // lanes 0-15: K = k0+0..15; lanes 16-31: K = k0+16..31
        v8bf lo = *(const v8bf*)(pA + k0 +     half * 16);
        v8bf hi = *(const v8bf*)(pA + k0 + 8 + half * 16);
        v16bf a = __builtin_shufflevector(lo, hi,
                   0,1,2,3,4,5,6,7,8,9,10,11,12,13,14,15);
        #pragma unroll
        for (int j = 0; j < 4; ++j) {
            // B fragment from LDS: lane = column, halves hold K lo/hi 16.
            const __bf16* pb = &ldsB[buf][(j * 16 + l) * 32 + half * 16];
            v16bf b = *(const v16bf*)pb;
            acc[j] = __builtin_amdgcn_wmma_f32_16x16x32_bf16(
                         false, a, false, b, (short)0, acc[j], false, false);
        }

        __syncthreads();                 // all reads of this panel done
    }

    if (active) {
        // C/D layout: lane l = column, VGPR v = row half*8+v
        #pragma unroll
        for (int j = 0; j < 4; ++j) {
            const int col = tn + j * 16 + l;
            const float bv = bias ? bias[col] : 0.0f;
            #pragma unroll
            for (int v = 0; v < 8; ++v) {
                const int row = tm * 16 + half * 8 + v;
                C[(size_t)row * N + col] = acc[j][v] + bv;
            }
        }
    }
}

// ---------------------------------------------------------------------------
// Conversion / utility kernels
// ---------------------------------------------------------------------------
__global__ void cast_bf16_k(const float* __restrict__ in, __bf16* __restrict__ out, int n) {
    int i = blockIdx.x * blockDim.x + threadIdx.x;
    if (i < n) out[i] = (__bf16)in[i];
}

// in: [R x C] row-major fp32  ->  out: [C x R] row-major bf16
__global__ void transpose_cast_bf16_k(const float* __restrict__ in, __bf16* __restrict__ out,
                                      int R, int Ccols) {
    int i = blockIdx.x * blockDim.x + threadIdx.x;
    if (i >= R * Ccols) return;
    int r = i / Ccols, c = i - r * Ccols;
    out[(size_t)c * R + r] = (__bf16)in[(size_t)r * Ccols + c];
}

__global__ void zero_f32_k(float* __restrict__ p, int n) {
    int i = blockIdx.x * blockDim.x + threadIdx.x;
    if (i < n) p[i] = 0.0f;
}

// ---------------------------------------------------------------------------
// Leaves: no children -> iou = bs, c = i*u, h = o*tanh(c)
// ---------------------------------------------------------------------------
__global__ void leaves_k(const float* __restrict__ XW, const float* __restrict__ bs,
                         float* __restrict__ cbuf, float* __restrict__ hbuf, int nLeaves) {
    int i = blockIdx.x * blockDim.x + threadIdx.x;
    if (i >= nLeaves * DM) return;
    int n = i >> 9, d = i & (DM - 1);
    const float* xw = XW + (size_t)n * (4 * DM);
    float iv = sigmoidf_(xw[d]          + bs[d]);
    float ov = sigmoidf_(xw[2 * DM + d] + bs[DM + d]);
    float uv = tanhf    (xw[3 * DM + d] + bs[2 * DM + d]);
    float cv = iv * uv;
    cbuf[(size_t)n * DM + d] = cv;
    hbuf[(size_t)n * DM + d] = ov * tanhf(cv);
}

// ---------------------------------------------------------------------------
// Per-level gather kernels (produce zero-padded bf16 GEMM inputs)
// ---------------------------------------------------------------------------
__global__ void gather_hsum_k(const float* __restrict__ hbuf, const int* __restrict__ children,
                              __bf16* __restrict__ hsumA, int base, int M, int Mpad) {
    int i = blockIdx.x * blockDim.x + threadIdx.x;
    if (i >= Mpad * DM) return;
    int m = i >> 9, d = i & (DM - 1);
    float s = 0.0f;
    if (m < M) {
        const int* ch = children + (size_t)(base + m) * KCH;
        #pragma unroll
        for (int k = 0; k < KCH; ++k) s += hbuf[(size_t)ch[k] * DM + d];
    }
    hsumA[(size_t)m * DM + d] = (__bf16)s;
}

__global__ void gather_ch_k(const float* __restrict__ hbuf, const int* __restrict__ children,
                            __bf16* __restrict__ CH, int base, int M, int Rpad) {
    int i = blockIdx.x * blockDim.x + threadIdx.x;
    if (i >= Rpad * DM) return;
    int r = i >> 9, d = i & (DM - 1);
    float v = 0.0f;
    if (r < KCH * M) {
        int m = r >> 2, k = r & 3;
        int ch = children[(size_t)(base + m) * KCH + k];
        v = hbuf[(size_t)ch * DM + d];
    }
    CH[(size_t)r * DM + d] = (__bf16)v;
}

// ---------------------------------------------------------------------------
// Combine: gates + cell update for one internal level
//   iou[m] = hsum@Ws + bs (from GEMM), F[4m+k] = ch_k@Wf + bf (from GEMM)
// ---------------------------------------------------------------------------
__global__ void combine_k(const float* __restrict__ XW, const float* __restrict__ iou,
                          const float* __restrict__ F, const int* __restrict__ children,
                          float* __restrict__ cbuf, float* __restrict__ hbuf,
                          int base, int M) {
    int i = blockIdx.x * blockDim.x + threadIdx.x;
    if (i >= M * DM) return;
    int m = i >> 9, d = i & (DM - 1);
    int n = base + m;
    const float* xw = XW + (size_t)n * (4 * DM);
    float ix = xw[d], fx = xw[DM + d], ox = xw[2 * DM + d], ux = xw[3 * DM + d];
    const float* io = iou + (size_t)m * (3 * DM);
    float iv = sigmoidf_(ix + io[d]);
    float ov = sigmoidf_(ox + io[DM + d]);
    float uv = tanhf    (ux + io[2 * DM + d]);
    float csum = 0.0f;
    const int* ch = children + (size_t)n * KCH;
    #pragma unroll
    for (int k = 0; k < KCH; ++k) {
        float fv = sigmoidf_(F[(size_t)(KCH * m + k) * DM + d] + fx);
        csum += fv * cbuf[(size_t)ch[k] * DM + d];
    }
    float cv = iv * uv + csum;
    cbuf[(size_t)n * DM + d] = cv;
    hbuf[(size_t)n * DM + d] = ov * tanhf(cv);
}

__global__ void copy_root_k(const float* __restrict__ hbuf, float* __restrict__ out) {
    int d = blockIdx.x * blockDim.x + threadIdx.x;
    if (d < DM) out[d] = hbuf[(size_t)(NN - 1) * DM + d];
}

// ---------------------------------------------------------------------------
// Host launcher
// ---------------------------------------------------------------------------
static inline unsigned cdivu(unsigned a, unsigned b) { return (a + b - 1) / b; }
static inline int ceil16(int x) { return (x + 15) & ~15; }
static inline unsigned gemm_grid(int Mtiles, int Nstrips) {
    return (unsigned)(((Mtiles + 7) / 8) * Nstrips);
}

extern "C" void kernel_launch(void* const* d_in, const int* in_sizes, int n_in,
                              void* d_out, int out_size, void* d_ws, size_t ws_size,
                              hipStream_t stream) {
    (void)in_sizes; (void)n_in; (void)out_size; (void)ws_size;
    const float* inputs   = (const float*)d_in[0];
    const float* Wx       = (const float*)d_in[1];
    const float* bx       = (const float*)d_in[2];
    const float* Ws       = (const float*)d_in[3];
    const float* bs       = (const float*)d_in[4];
    const float* Wf       = (const float*)d_in[5];
    const float* bf       = (const float*)d_in[6];
    const int*   children = (const int*)d_in[7];
    float* out = (float*)d_out;

    // ---- workspace carve-up (256B aligned) ----
    char*  ws  = (char*)d_ws;
    size_t off = 0;
    auto take = [&](size_t bytes) -> char* {
        char* p = ws + off;
        off = (off + bytes + 255) & ~(size_t)255;
        return p;
    };
    __bf16* A_in  = (__bf16*)take((size_t)NN * DM * 2);          // inputs, bf16
    __bf16* WxT   = (__bf16*)take((size_t)4 * DM * DM * 2);      // [2048 x 512]
    __bf16* WsT   = (__bf16*)take((size_t)3 * DM * DM * 2);      // [1536 x 512]
    __bf16* WfT   = (__bf16*)take((size_t)DM * DM * 2);          // [512 x 512]
    float*  XW    = (float*)take((size_t)NN * 4 * DM * 4);       // [4096 x 2048]
    float*  hbuf  = (float*)take((size_t)(NN + 1) * DM * 4);     // sentinel row NN
    float*  cbuf  = (float*)take((size_t)(NN + 1) * DM * 4);
    __bf16* hsumA = (__bf16*)take((size_t)688 * DM * 2);         // ceil16(683)=688
    __bf16* CH    = (__bf16*)take((size_t)2736 * DM * 2);        // ceil16(4*683)=2736
    float*  iou   = (float*)take((size_t)688 * 3 * DM * 4);
    float*  F     = (float*)take((size_t)2736 * DM * 4);

    const int TB = 256;

    // ---- phase 0: precision conversion + weight transposes ----
    cast_bf16_k<<<cdivu(NN * DM, TB), TB, 0, stream>>>(inputs, A_in, NN * DM);
    transpose_cast_bf16_k<<<cdivu(DM * 4 * DM, TB), TB, 0, stream>>>(Wx, WxT, DM, 4 * DM);
    transpose_cast_bf16_k<<<cdivu(DM * 3 * DM, TB), TB, 0, stream>>>(Ws, WsT, DM, 3 * DM);
    transpose_cast_bf16_k<<<cdivu(DM * DM, TB), TB, 0, stream>>>(Wf, WfT, DM, DM);
    zero_f32_k<<<cdivu(DM, TB), TB, 0, stream>>>(hbuf + (size_t)NN * DM, DM);
    zero_f32_k<<<cdivu(DM, TB), TB, 0, stream>>>(cbuf + (size_t)NN * DM, DM);

    // ---- phase 1: XW = inputs @ Wx + bx   (4096x2048, K=512) ----
    wmma_gemm_bf16nt<<<gemm_grid(NN / 16, (4 * DM) / 64), TB, 0, stream>>>(
        A_in, WxT, bx, XW, NN / 16, (4 * DM) / 64, DM, 4 * DM);

    // ---- phase 2: leaves (ids 0..3071), fully parallel ----
    leaves_k<<<cdivu(3072 * DM, TB), TB, 0, stream>>>(XW, bs, cbuf, hbuf, 3072);

    // ---- phase 3: internal levels bottom-up ----
    struct Lv { int base, M; };
    const Lv levels[6] = { {3072, 683}, {3755, 256}, {4011, 64},
                           {4075, 16},  {4091, 4},   {4095, 1} };
    for (int li = 0; li < 6; ++li) {
        const int base = levels[li].base, M = levels[li].M;
        const int Mp = ceil16(M);                 // rows for hsum GEMM
        const int Rp = ceil16(KCH * M);           // rows for per-child GEMM

        gather_hsum_k<<<cdivu(Mp * DM, TB), TB, 0, stream>>>(hbuf, children, hsumA, base, M, Mp);
        gather_ch_k  <<<cdivu(Rp * DM, TB), TB, 0, stream>>>(hbuf, children, CH,    base, M, Rp);

        // iou[Mp x 1536] = hsum @ Ws + bs
        wmma_gemm_bf16nt<<<gemm_grid(Mp / 16, (3 * DM) / 64), TB, 0, stream>>>(
            hsumA, WsT, bs, iou, Mp / 16, (3 * DM) / 64, DM, 3 * DM);

        // F[Rp x 512] = CH @ Wf + bf
        wmma_gemm_bf16nt<<<gemm_grid(Rp / 16, DM / 64), TB, 0, stream>>>(
            CH, WfT, bf, F, Rp / 16, DM / 64, DM, DM);

        combine_k<<<cdivu(M * DM, TB), TB, 0, stream>>>(XW, iou, F, children,
                                                        cbuf, hbuf, base, M);
    }

    // ---- phase 4: emit root hidden state ----
    copy_root_k<<<cdivu(DM, TB), TB, 0, stream>>>(hbuf, out);
}
